// GraphConvLayer_55645596287598
// MI455X (gfx1250) — matrix operand order repaired and checked by reading, not compiled, not run on hardware
//
#include <hip/hip_runtime.h>
#include <hip/hip_bf16.h>

#define IN_DIM   512
#define CONC     1536
#define HIDDEN   1024
#define OUT_DIM  512
#define N_OBJ    50000
#define N_PRED   100000
#define MT       32            // edge rows per block
#define NBLOCKS  (N_PRED / MT) // 3125

typedef __attribute__((ext_vector_type(16))) __bf16          v16bf;
typedef __attribute__((ext_vector_type(8)))  float           v8f;
typedef __attribute__((ext_vector_type(8)))  unsigned short  us8;

union FragU {
    us8   h[2];
    v16bf b;
};

static __device__ __forceinline__ unsigned short f2bf(float f) {
    unsigned int u = __builtin_bit_cast(unsigned int, f);
    return (unsigned short)((u + 0x7FFFu + ((u >> 16) & 1u)) >> 16);
}

static __device__ __forceinline__ v8f wmma_bf16(v16bf a, v16bf b, v8f c) {
    // D = A(16x32 bf16) * B(32x16 bf16) + C(16x16 f32)
    return __builtin_amdgcn_wmma_f32_16x16x32_bf16(
        /*neg_a=*/false, a, /*neg_b=*/false, b,
        /*c_mod=*/(short)0, c, /*reuse_a=*/false, /*reuse_b=*/false);
}

// One K=32 step for two 16-row m-tiles sharing NT B-fragments.
template <int NT>
static __device__ __forceinline__ void gemm_step(
    const unsigned short* aptr0, const unsigned short* aptr1,
    const unsigned short* __restrict__ bp,
    v8f (&acc0)[NT], v8f (&acc1)[NT]) {
    FragU a0, a1;
    a0.h[0] = *(const us8*)(aptr0);
    a0.h[1] = *(const us8*)(aptr0 + 16);
    a1.h[0] = *(const us8*)(aptr1);
    a1.h[1] = *(const us8*)(aptr1 + 16);
    #pragma unroll
    for (int t = 0; t < NT; ++t) {
        FragU b;
        #pragma unroll
        for (int s = 0; s < 8; ++s) {
            b.h[0][s] = bp[(long long)s * (NT * 128) + t * 16];
            b.h[1][s] = bp[(long long)(16 + s) * (NT * 128) + t * 16];
        }
        acc0[t] = wmma_bf16(a0.b, b.b, acc0[t]);
        acc1[t] = wmma_bf16(a1.b, b.b, acc1[t]);
    }
}

// ---------------------------------------------------------------- utility ---
__global__ void zero_f32(float* p, long long n) {
    long long i = (long long)blockIdx.x * blockDim.x + threadIdx.x;
    long long stride = (long long)gridDim.x * blockDim.x;
    for (; i < n; i += stride) p[i] = 0.0f;
}

__global__ void cvt_f32_bf16(const float* __restrict__ src,
                             unsigned short* __restrict__ dst, long long n4) {
    long long i = (long long)blockIdx.x * blockDim.x + threadIdx.x;
    long long stride = (long long)gridDim.x * blockDim.x;
    for (; i < n4; i += stride) {
        float4 v = *(const float4*)(src + 4 * i);
        uint2 p;
        p.x = (unsigned int)f2bf(v.x) | ((unsigned int)f2bf(v.y) << 16);
        p.y = (unsigned int)f2bf(v.z) | ((unsigned int)f2bf(v.w) << 16);
        *(uint2*)(dst + 4 * i) = p;
    }
}

__global__ void count_edges(const int* __restrict__ edges, float* __restrict__ counts) {
    int e = blockIdx.x * blockDim.x + threadIdx.x;
    if (e < N_PRED) {
        atomicAdd(counts + edges[2 * e + 0], 1.0f);
        atomicAdd(counts + edges[2 * e + 1], 1.0f);
    }
}

__global__ void finalize_div(float* __restrict__ out_obj, const float* __restrict__ counts) {
    long long i = (long long)blockIdx.x * blockDim.x + threadIdx.x;
    long long stride = (long long)gridDim.x * blockDim.x;
    long long n = (long long)N_OBJ * OUT_DIM;
    for (; i < n; i += stride) {
        float c = counts[i >> 9];  // OUT_DIM == 512
        out_obj[i] *= (c < 1.0f) ? 1.0f : (1.0f / c);
    }
}

// ------------------------------------------------------------- fused MLPs ---
struct FusedParams {
    const float* obj;
    const float* pred;
    const int*   edges;
    const unsigned short* w1[3];   // 1536x1024 bf16, row-major
    const unsigned short* w2[3];   // 1024x512  bf16, row-major
    const float* b1[3];
    const float* b2[3];
    float* out_obj;                // 50000x512, accumulated via atomics (g=1,2)
    float* out_pred;               // 100000x512, direct NT store (g=0)
};

__global__ __launch_bounds__(256)
void fused_graph_mlp(FusedParams p) {
    __shared__ unsigned short lds[MT * HIDDEN];  // 64 KB: A tile, then H tile

    const int tid  = threadIdx.x;
    const int lane = tid & 31;
    const int wv   = tid >> 5;        // 8 waves
    const int e0   = blockIdx.x * MT;

    const int m_lo   = lane & 15;          // row within 16-row m-tile
    const int khalf  = (lane >> 4) * 8;    // A-frag K sub-offset per ISA layout
    const int mrow0  = (lane >> 4) * 8;    // C/D layout: m = v + 8*(lane>=16)
    const int ncol   = lane & 15;          // C/D layout: n = lane&15

    for (int g = 0; g < 3; ++g) {
        // ---- load gathered/concat A tile (unique 1024 cols) as bf16 -------
        for (int i = tid; i < MT * HIDDEN / 4; i += 256) {
            int m  = i >> 8;              // 256 float4 slots per row
            int c  = (i & 255) * 4;
            int e  = e0 + m;
            const float* src;
            if (c < IN_DIM) {
                int o = p.edges[2 * e + 1];
                src = p.obj + (long long)o * IN_DIM + c;
            } else {
                src = p.pred + (long long)e * IN_DIM + (c - IN_DIM);
            }
            float4 v = *(const float4*)src;
            uint2 pk;
            pk.x = (unsigned int)f2bf(v.x) | ((unsigned int)f2bf(v.y) << 16);
            pk.y = (unsigned int)f2bf(v.z) | ((unsigned int)f2bf(v.w) << 16);
            *(uint2*)&lds[m * HIDDEN + c] = pk;
        }
        __syncthreads();

        // ---- stage 1: H = relu(A(32x1536) @ W1(1536x1024) + b1) -----------
        // Logical A = [obj(512) | pred(512) | obj(512)]; LDS holds the unique
        // 1024 cols. Split K loop so each range has a linear LDS induction
        // (no per-iter wrap select in the hot loop).
        v8f acc0[8], acc1[8];
        #pragma unroll
        for (int t = 0; t < 8; ++t) { acc0[t] = (v8f)0.0f; acc1[t] = (v8f)0.0f; }

        const unsigned short* abase = &lds[m_lo * HIDDEN + khalf];
        const unsigned short* w1p   = p.w1[g] + (long long)khalf * HIDDEN + wv * 128 + m_lo;

        for (int k0 = 0; k0 < HIDDEN; k0 += 32)
            gemm_step<8>(abase + k0, abase + k0 + 16 * HIDDEN,
                         w1p + (long long)k0 * HIDDEN, acc0, acc1);
        for (int k0 = HIDDEN; k0 < CONC; k0 += 32)
            gemm_step<8>(abase + (k0 - HIDDEN), abase + (k0 - HIDDEN) + 16 * HIDDEN,
                         w1p + (long long)k0 * HIDDEN, acc0, acc1);
        __syncthreads();   // everyone done reading A before H overwrites it

        // ---- bias + relu, write H tile (32x1024 bf16) ---------------------
        float bias1[8];
        #pragma unroll
        for (int t = 0; t < 8; ++t) bias1[t] = p.b1[g][wv * 128 + t * 16 + ncol];
        #pragma unroll
        for (int t = 0; t < 8; ++t) {
            int n = wv * 128 + t * 16 + ncol;
            #pragma unroll
            for (int v = 0; v < 8; ++v) {
                float x0 = acc0[t][v] + bias1[t];
                float x1 = acc1[t][v] + bias1[t];
                lds[(mrow0 + v) * HIDDEN + n]      = f2bf(x0 > 0.0f ? x0 : 0.0f);
                lds[(16 + mrow0 + v) * HIDDEN + n] = f2bf(x1 > 0.0f ? x1 : 0.0f);
            }
        }
        __syncthreads();

        // ---- stage 2: OUT = H(32x1024) @ W2(1024x512) + b2 ----------------
        v8f acc2a[4], acc2b[4];
        #pragma unroll
        for (int t = 0; t < 4; ++t) { acc2a[t] = (v8f)0.0f; acc2b[t] = (v8f)0.0f; }

        const unsigned short* w2p = p.w2[g] + (long long)khalf * OUT_DIM + wv * 64 + m_lo;
        for (int k0 = 0; k0 < HIDDEN; k0 += 32)
            gemm_step<4>(abase + k0, abase + k0 + 16 * HIDDEN,
                         w2p + (long long)k0 * OUT_DIM, acc2a, acc2b);

        // ---- emit ---------------------------------------------------------
        float bias2[4];
        #pragma unroll
        for (int t = 0; t < 4; ++t) bias2[t] = p.b2[g][wv * 64 + t * 16 + ncol];

        if (g == 0) {
            #pragma unroll
            for (int mt = 0; mt < 2; ++mt)
                #pragma unroll
                for (int v = 0; v < 8; ++v) {
                    long long r = e0 + mt * 16 + mrow0 + v;
                    float* dst = p.out_pred + r * OUT_DIM + wv * 64 + ncol;
                    #pragma unroll
                    for (int t = 0; t < 4; ++t) {
                        float val = (mt ? acc2b[t][v] : acc2a[t][v]) + bias2[t];
                        __builtin_nontemporal_store(val, dst + t * 16);
                    }
                }
        } else {
            const int sel = (g == 1) ? 0 : 1;
            #pragma unroll
            for (int mt = 0; mt < 2; ++mt)
                #pragma unroll
                for (int v = 0; v < 8; ++v) {
                    int r = e0 + mt * 16 + mrow0 + v;
                    long long idx = p.edges[2 * r + sel];
                    float* dst = p.out_obj + idx * OUT_DIM + wv * 64 + ncol;
                    #pragma unroll
                    for (int t = 0; t < 4; ++t) {
                        float val = (mt ? acc2b[t][v] : acc2a[t][v]) + bias2[t];
                        atomicAdd(dst + t * 16, val);
                    }
                }
        }
        __syncthreads();   // before next MLP reloads the A tile
    }
}

// ------------------------------------------------------------------ launch --
extern "C" void kernel_launch(void* const* d_in, const int* in_sizes, int n_in,
                              void* d_out, int out_size, void* d_ws, size_t ws_size,
                              hipStream_t stream) {
    const float* obj   = (const float*)d_in[0];
    const float* pred  = (const float*)d_in[1];
    const int*   edges = (const int*)d_in[2];

    const long long W1E = (long long)CONC * HIDDEN;     // 1,572,864
    const long long W2E = (long long)HIDDEN * OUT_DIM;  //   524,288

    unsigned short* wbf = (unsigned short*)d_ws;
    unsigned short* w1bf[3], *w2bf[3];
    unsigned short* cur = wbf;
    for (int g = 0; g < 3; ++g) {
        w1bf[g] = cur; cur += W1E;
        w2bf[g] = cur; cur += W2E;
    }
    float* counts = (float*)cur;  // 50000 floats after 12.58 MB of bf16 weights

    float* out_obj  = (float*)d_out;                               // 50000x512
    float* out_pred = (float*)d_out + (long long)N_OBJ * OUT_DIM;  // 100000x512

    // 1) zero accumulation targets (harness poisons buffers with 0xAA)
    zero_f32<<<2048, 256, 0, stream>>>(out_obj, (long long)N_OBJ * OUT_DIM);
    zero_f32<<<64, 256, 0, stream>>>(counts, N_OBJ);

    // 2) weights f32 -> bf16 (L2-resident, reused by 3125 blocks)
    for (int g = 0; g < 3; ++g) {
        cvt_f32_bf16<<<1024, 256, 0, stream>>>((const float*)d_in[3 + 4 * g],
                                               w1bf[g], W1E / 4);
        cvt_f32_bf16<<<1024, 256, 0, stream>>>((const float*)d_in[5 + 4 * g],
                                               w2bf[g], W2E / 4);
    }

    // 3) degree counts
    count_edges<<<(N_PRED + 255) / 256, 256, 0, stream>>>(edges, counts);

    // 4) fused gather + 3x(GEMM-ReLU-GEMM) + scatter
    FusedParams p;
    p.obj = obj; p.pred = pred; p.edges = edges;
    for (int g = 0; g < 3; ++g) {
        p.w1[g] = w1bf[g];
        p.w2[g] = w2bf[g];
        p.b1[g] = (const float*)d_in[4 + 4 * g];
        p.b2[g] = (const float*)d_in[6 + 4 * g];
    }
    p.out_obj = out_obj; p.out_pred = out_pred;
    fused_graph_mlp<<<NBLOCKS, 256, 0, stream>>>(p);

    // 5) mean-normalize object outputs
    finalize_div<<<2048, 256, 0, stream>>>(out_obj, counts);
}
